// AdaptiveSmoothing_23630910063235
// MI455X (gfx1250) — matrix operand order repaired and checked by reading, not compiled, not run on hardware
//
#include <hip/hip_runtime.h>
#include <math.h>

// ---------------- CDNA5 vector types ----------------
typedef __attribute__((ext_vector_type(16))) __bf16         v16bf;
typedef __attribute__((ext_vector_type(8)))  float          v8f;
typedef __attribute__((ext_vector_type(8)))  unsigned int   v8u;
typedef __attribute__((ext_vector_type(4)))  unsigned int   u32x4;
typedef __attribute__((ext_vector_type(8)))  int            i32x8;
typedef __attribute__((ext_vector_type(4)))  int            i32x4;

// ---------------- Problem constants ----------------
#define SIZE_T   12
#define SIZE_X   10
#define KH       21            // taps along H (dx)
#define KW       25            // taps along W (dt)
#define DT_STEP  5.0f
#define DX_STEP  0.1f

// K-blocking: one dx-row per K-block; K=32 = 25 dt-taps + 7 zero cols in B.
#define NKB      KH            // 21

#define TILE_W   16            // output pixels per wave (M dim)
#define ROWS_PER_BLK 8         // 8 waves per block, one output row each
#define HALO_H   (ROWS_PER_BLK + KH - 1)   // 28
#define HALO_W   (TILE_W + KW - 1)         // 40  (f32 staging tile width)
#define PITCH    48                        // interleaved-tile pitch (covers col m+31 = 46)

#define BATCH 8
#define DIM_H 512
#define DIM_W 4096

#if defined(__has_builtin)
#  if __has_builtin(__builtin_amdgcn_tensor_load_to_lds) && \
      __has_builtin(__builtin_amdgcn_s_wait_tensorcnt)
#    define HAVE_TDM 1
#  endif
#endif
#ifndef HAVE_TDM
#  define HAVE_TDM 0
#endif

// float -> bf16, round-to-nearest-even (no NaNs reach this)
__device__ __forceinline__ unsigned short f2bf(float f) {
    unsigned int u = __float_as_uint(f);
    unsigned int r = u + 0x7FFFu + ((u >> 16) & 1u);
    return (unsigned short)(r >> 16);
}

// ================================================================
// Setup kernel: per-lane B fragments for v_wmma_f32_16x16x32_bf16.
// B is (K=32) x (N=16); K = dt index (25 taps, 7 zero-pad), block kb
// selects the dx row.  Columns n=0 (k_cong), n=1 (k_free); rest 0.
// Lane layout: n = lane&15, half = lane>>4; element (2g+e) holds
// K = 2g + e + 16*half.  Fragment = 16 bf16 = 32 B per lane.
// ================================================================
__global__ void asm5_build_bfrag(const float* __restrict__ delta,
                                 const float* __restrict__ tau,
                                 const float* __restrict__ c_cong,
                                 const float* __restrict__ c_free,
                                 unsigned short* __restrict__ wsB) {
    const int kb   = blockIdx.x;      // dx row, 0..KH-1
    const int lane = threadIdx.x;     // 0..31
    const int half = lane >> 4;
    const int n    = lane & 15;

    const float inv_tau   = 1.0f / *tau;
    const float inv_delta = 1.0f / *delta;
    const float sc = 3600.0f / *c_cong;
    const float sf = 3600.0f / *c_free;

    unsigned short* dst = wsB + (size_t)(kb * 32 + lane) * 16;
#pragma unroll
    for (int g = 0; g < 8; ++g) {
#pragma unroll
        for (int e = 0; e < 2; ++e) {
            const int dt = 2 * g + e + 16 * half;   // K index == dt tap
            float w = 0.0f;
            if (dt < KW && n < 2) {
                const float X = (float)(kb - SIZE_X) * DX_STEP;
                const float T = (float)(dt - SIZE_T) * DT_STEP;
                const float s = (n == 0) ? sc : sf;
                const float tt = T - X * s;        // k(-X,-T) == k(X,T)
                w = expf(-(fabsf(tt) * inv_tau + fabsf(X) * inv_delta));
            }
            dst[2 * g + e] = f2bf(w);
        }
    }
}

// ================================================================
// Main kernel: implicit-GEMM convolution on the bf16 WMMA pipe.
// Block = 256 threads = 8 waves; tile = 8 rows x 16 cols of output.
// Interior blocks stage the f32 halo tile via the Tensor Data Mover;
// boundary blocks stage with guarded vector loads.  One shared pass
// converts to an interleaved {bf16 mask | bf16 data} u32 tile that
// feeds aligned ds_load_b32 + v_perm_b32 A-fragment builds.
// ================================================================
__global__ __launch_bounds__(256) void asm5_conv_wmma(
        const float* __restrict__ raw,
        const unsigned short* __restrict__ wsB,
        const float* __restrict__ v_thr_p,
        const float* __restrict__ v_delta_p,
        float* __restrict__ out) {

    __shared__ float        sRawF32[HALO_H * HALO_W];   // 4480 B
    __shared__ unsigned int sDM[HALO_H * PITCH];        // 5376 B, {mask|data}
    __shared__ float        sScr[ROWS_PER_BLK][TILE_W][4];

    const int b   = blockIdx.z;
    const int i0  = blockIdx.y * ROWS_PER_BLK;
    const int j0  = blockIdx.x * TILE_W;
    const int tid = threadIdx.x;

    const float* src = raw + (size_t)b * DIM_H * DIM_W;

#if HAVE_TDM
    const bool interior = (i0 >= SIZE_X) && (i0 + ROWS_PER_BLK + SIZE_X <= DIM_H) &&
                          (j0 >= SIZE_T) && (j0 + TILE_W + SIZE_T <= DIM_W);
#else
    const bool interior = false;
#endif

#if HAVE_TDM
    if (interior) {
        if (tid < 32) {   // wave 0 drives the TDM; EXEC is ignored by TDM
            const unsigned long long ga = (unsigned long long)(uintptr_t)
                (src + (size_t)(i0 - SIZE_X) * DIM_W + (j0 - SIZE_T));
            const unsigned int ldsa = (unsigned int)(uintptr_t)&sRawF32[0];
            const unsigned int tdx  = (unsigned int)(DIM_W - (j0 - SIZE_T));
            const unsigned int tdy  = (unsigned int)(DIM_H - (i0 - SIZE_X));

            // D# group 0: count=1 | lds_addr | global_addr[56:0] | type=2
            u32x4 g0;
            g0[0] = 1u;
            g0[1] = ldsa;
            g0[2] = (unsigned int)(ga & 0xFFFFFFFFu);
            g0[3] = (unsigned int)((ga >> 32) & 0x01FFFFFFu) | 0x80000000u;

            // D# group 1: data_size=4B, tensor_dim0/1, tile 40x28, stride 4096
            i32x8 g1;
            g1[0] = (int)(2u << 16);                                   // data_size
            g1[1] = (int)((tdx & 0xFFFFu) << 16);                      // dim0 lo
            g1[2] = (int)(((tdx >> 16) & 0xFFFFu) | ((tdy & 0xFFFFu) << 16));
            g1[3] = (int)(((tdy >> 16) & 0xFFFFu) | ((unsigned)HALO_W << 16));
            g1[4] = (int)HALO_H;                                       // tile_dim1
            g1[5] = (int)DIM_W;                                        // dim0 stride lo
            g1[6] = 0;
            g1[7] = 0;

            const i32x4 z4 = {0, 0, 0, 0};
#if __clang_major__ >= 23
            const i32x8 z8 = {0, 0, 0, 0, 0, 0, 0, 0};
            __builtin_amdgcn_tensor_load_to_lds(g0, g1, z4, z4, z8, 0);
#else
            __builtin_amdgcn_tensor_load_to_lds(g0, g1, z4, z4, 0);
#endif
            __builtin_amdgcn_s_wait_tensorcnt(0);
        }
    } else
#endif
    {   // boundary (or no-TDM) staging: guarded loads, OOB -> 0.0f
        for (int e = tid; e < HALO_H * HALO_W; e += 256) {
            const int r  = e / HALO_W;
            const int c  = e - r * HALO_W;
            const int gi = i0 - SIZE_X + r;
            const int gj = j0 - SIZE_T + c;
            float v = 0.0f;
            if (gi >= 0 && gi < DIM_H && gj >= 0 && gj < DIM_W)
                v = src[(size_t)gi * DIM_W + gj];
            sRawF32[e] = v;
        }
    }
    __syncthreads();

    // Convert to interleaved {bf16 mask (hi) | bf16 data (lo)} tile.
    // Pad cols [HALO_W, PITCH) are zero (finite -> safe under B zero-pad).
    for (int e = tid; e < HALO_H * PITCH; e += 256) {
        const int r = e / PITCH;
        const int c = e - r * PITCH;
        unsigned int packed = 0u;
        if (c < HALO_W) {
            const float v = sRawF32[r * HALO_W + c];
            if (!__builtin_isnan(v))
                packed = (unsigned int)f2bf(v) | 0x3F800000u;  // mask=1.0bf16
        }
        sDM[e] = packed;
    }
    __syncthreads();

    const int wave = tid >> 5;        // output row within block
    const int lane = tid & 31;
    const int half = lane >> 4;
    const int m    = lane & 15;       // A-matrix row = pixel index

    // All A-element addresses: dmBase + (kb*PITCH + k0) with k0 immediate.
    const unsigned int* dmBase = &sDM[wave * PITCH + m];

    v8f cd = {};   // cols 0,1: sum_cong, sum_free
    v8f cm = {};   // cols 0,1: N_cong,   N_free
    const v8u* __restrict__ wsBv = (const v8u*)wsB;
    __builtin_prefetch(wsBv, 0, 1);   // global_prefetch_b8 on the B stream

#pragma unroll
    for (int kb = 0; kb < NKB; ++kb) {
        const v8u braw = wsBv[kb * 32 + lane];
        v8u ad, am;   // A fragments as 8 dwords (2 bf16 each)
#pragma unroll
        for (int g = 0; g < 8; ++g) {
            // element pair (2g, 2g+1): K = k0, k0+1 (== dt taps)
            const int k0 = (g << 1) | (half << 4);
            const unsigned int p0 = dmBase[kb * PITCH + k0];
            const unsigned int p1 = dmBase[kb * PITCH + k0 + 1];
            ad[g] = __builtin_amdgcn_perm(p1, p0, 0x05040100u);  // data pair
            am[g] = __builtin_amdgcn_perm(p1, p0, 0x07060302u);  // mask pair
        }
        const v16bf a_d = __builtin_bit_cast(v16bf, ad);
        const v16bf a_m = __builtin_bit_cast(v16bf, am);
        const v16bf bb  = __builtin_bit_cast(v16bf, braw);
        cd = __builtin_amdgcn_wmma_f32_16x16x32_bf16(false, a_d, false, bb,
                                                     (short)0, cd, false, false);
        cm = __builtin_amdgcn_wmma_f32_16x16x32_bf16(false, a_m, false, bb,
                                                     (short)0, cm, false, false);
    }

    // D layout: lane n=lane&15 holds rows M = vgpr + 8*half of column n.
    float* scr = &sScr[wave][0][0];
    const int ncol = lane & 15;
    if (ncol < 2) {
#pragma unroll
        for (int v = 0; v < 8; ++v) {
            const int row = v + 8 * half;
            scr[row * 4 + ncol]     = cd[v];   // sum_cong / sum_free
            scr[row * 4 + 2 + ncol] = cm[v];   // N_cong   / N_free
        }
    }
    __syncthreads();

    if (lane < 16) {
        const float sumc = scr[lane * 4 + 0];
        const float sumf = scr[lane * 4 + 1];
        const float Nc   = scr[lane * 4 + 2];
        const float Nf   = scr[lane * 4 + 3];
        const bool hc = Nc > 0.0f;
        const bool hf = Nf > 0.0f;
        const float vc = hc ? sumc / (Nc + 1e-8f) : 0.0f;
        const float vf = hf ? sumf / (Nf + 1e-8f) : 0.0f;
        const float vmin = fminf(vc, vf);
        const float w = 0.5f * (1.0f + tanhf((*v_thr_p - vmin) / *v_delta_p));
        const float vmix = w * vc + (1.0f - w) * vf;
        float v = (hc && hf) ? vmix : (hc ? vc : vf);
        if (!(hc || hf)) v = __builtin_nanf("");
        out[((size_t)b * DIM_H + (i0 + wave)) * DIM_W + j0 + lane] = v;
    }
}

extern "C" void kernel_launch(void* const* d_in, const int* in_sizes, int n_in,
                              void* d_out, int out_size, void* d_ws, size_t ws_size,
                              hipStream_t stream) {
    (void)in_sizes; (void)n_in; (void)out_size; (void)ws_size;
    const float* raw     = (const float*)d_in[0];
    const float* delta   = (const float*)d_in[1];
    const float* tau     = (const float*)d_in[2];
    const float* c_cong  = (const float*)d_in[3];
    const float* c_free  = (const float*)d_in[4];
    const float* v_thr   = (const float*)d_in[5];
    const float* v_delta = (const float*)d_in[6];
    float* out = (float*)d_out;
    unsigned short* wsB = (unsigned short*)d_ws;   // NKB*32*16 bf16 = 21504 B

    asm5_build_bfrag<<<dim3(NKB), dim3(32), 0, stream>>>(delta, tau, c_cong,
                                                         c_free, wsB);

    dim3 grid(DIM_W / TILE_W, DIM_H / ROWS_PER_BLK, BATCH);
    asm5_conv_wmma<<<grid, dim3(256), 0, stream>>>(raw, wsB, v_thr, v_delta, out);
}
